// Decompose_43396349558932
// MI455X (gfx1250) — matrix operand order repaired and checked by reading, not compiled, not run on hardware
//
#include <hip/hip_runtime.h>

// Decompose kernel for MI455X (gfx1250, wave32, WMMA).
//
// Roofline: 34.4 GFLOP vs ~0.77 GB of HBM traffic -> 33us memory floor at
// 23.3 TB/s. fp32 WMMA (K=4) would be compute-bound; bf16 hi/lo split with
// 3x wmma_f32_16x16x32_bf16 per K=32 slab is ~fp32-accurate and makes the
// kernel memory-bound.

#define C_DIM 8
#define E_DIM 128
#define O_DIM 256     // 2*E
#define LDS_STRIDE 136  // 128 + 8 bf16 pad -> conflict-free ds_read_b128

typedef __bf16 bf16_t;
typedef __attribute__((ext_vector_type(16))) bf16_t          v16bf;
typedef __attribute__((ext_vector_type(8)))  float           v8f;
typedef __attribute__((ext_vector_type(4)))  float           v4f;
typedef __attribute__((ext_vector_type(8)))  unsigned short  u16x8;
typedef __attribute__((ext_vector_type(16))) unsigned short  u16x16;

// fp32 -> bf16 with round-to-nearest-even (bit-exact, no __bf16 arithmetic).
__device__ __forceinline__ unsigned short f2bf(float f) {
  unsigned int u = __builtin_bit_cast(unsigned int, f);
  u += 0x7FFFu + ((u >> 16) & 1u);
  return (unsigned short)(u >> 16);
}

// Glue two 8x16-bit chunks into the 16-element WMMA operand vector.
__device__ __forceinline__ v16bf mk16(u16x8 a, u16x8 b) {
  u16x16 c = __builtin_shufflevector(a, b, 0, 1, 2, 3, 4, 5, 6, 7,
                                           8, 9, 10, 11, 12, 13, 14, 15);
  union { u16x16 u; v16bf v; } p;
  p.u = c;
  return p.v;
}

// ---------------------------------------------------------------------------
// Kernel 1: split W (256x128 fp32) into bf16 hi/lo planes in workspace.
// ---------------------------------------------------------------------------
__global__ __launch_bounds__(256) void wprep_kernel(
    const float* __restrict__ W,
    unsigned short* __restrict__ Whi,
    unsigned short* __restrict__ Wlo) {
  const int i = blockIdx.x * 256 + threadIdx.x;
  if (i < O_DIM * E_DIM) {
    const float x = W[i];
    const unsigned short h = f2bf(x);
    const float hf = __builtin_bit_cast(float, (unsigned int)h << 16);
    Whi[i] = h;
    Wlo[i] = f2bf(x - hf);
  }
}

// ---------------------------------------------------------------------------
// Kernel 2: block = 8 wave32, 32 A-rows (4 samples) x 256 outputs.
// Wave w: M-tile (w&1), N-tiles (w>>1)+4*{0..3}. K=128 in four K=32 slabs,
// 3 bf16 WMMAs per slab per N-tile (hi*hi + hi*lo + lo*hi).
// ---------------------------------------------------------------------------
__global__ __launch_bounds__(256) void decompose_gemm_kernel(
    const float* __restrict__ feat,
    const unsigned short* __restrict__ Whi,
    const unsigned short* __restrict__ Wlo,
    const float* __restrict__ bias,
    float* __restrict__ out) {
  __shared__ __align__(16) unsigned short sAh[32 * LDS_STRIDE];
  __shared__ __align__(16) unsigned short sAl[32 * LDS_STRIDE];

  const int t   = threadIdx.x;
  const int blk = blockIdx.x;

  // ---- Phase 1: load 32x128 fp32 A tile, split to bf16 hi/lo in LDS ----
  {
    const int lr = t >> 3;            // local A-row 0..31
    const int cf = (t & 7) << 4;      // 16 floats per thread
    // local row lr -> sample k = 4*blk + lr/8 -> feat row 2k, channel lr%8
    const float* src = feat
        + (size_t)(8 * blk + 2 * (lr >> 3)) * (size_t)(C_DIM * E_DIM)
        + (size_t)((lr & 7) * E_DIM + cf);
    float xv[16];
#pragma unroll
    for (int q = 0; q < 4; ++q) {
      v4f fq = __builtin_nontemporal_load(((const v4f*)src) + q);
#pragma unroll
      for (int j = 0; j < 4; ++j) xv[4 * q + j] = fq[j];
    }
    u16x8 h[2], l[2];
#pragma unroll
    for (int i = 0; i < 16; ++i) {
      const unsigned short hb = f2bf(xv[i]);
      const float hf = __builtin_bit_cast(float, (unsigned int)hb << 16);
      h[i >> 3][i & 7] = hb;
      l[i >> 3][i & 7] = f2bf(xv[i] - hf);
    }
    unsigned short* dh = &sAh[lr * LDS_STRIDE + cf];
    unsigned short* dl = &sAl[lr * LDS_STRIDE + cf];
    *(u16x8*)(dh)     = h[0];
    *(u16x8*)(dh + 8) = h[1];
    *(u16x8*)(dl)     = l[0];
    *(u16x8*)(dl + 8) = l[1];
  }
  __syncthreads();

  // ---- Phase 2: WMMA main loop ----
  const int lane = t & 31;
  const int w    = t >> 5;
  const int mt   = w & 1;        // M-tile within block
  const int ng   = w >> 1;       // N-tile group 0..3
  const int g    = lane >> 4;    // lane half
  const int ln   = lane & 15;

  v8f acc[4] = {};

  // A fragment (contiguous k order, matching B): lane half g holds the 16
  // contiguous K values g*16..g*16+15 of its row.
  const unsigned short* pAh = &sAh[(mt * 16 + ln) * LDS_STRIDE + g * 16];
  const unsigned short* pAl = &sAl[(mt * 16 + ln) * LDS_STRIDE + g * 16];

#pragma unroll
  for (int kk = 0; kk < 4; ++kk) {
    const int kb = kk * 32;
    v16bf Ah = mk16(*(const u16x8*)(pAh + kb), *(const u16x8*)(pAh + kb + 8));
    v16bf Al = mk16(*(const u16x8*)(pAl + kb), *(const u16x8*)(pAl + kb + 8));
#pragma unroll
    for (int i = 0; i < 4; ++i) {
      const int nt = ng + 4 * i;
      // B fragment (contiguous k order, matching A): lane half g holds the 16
      // contiguous K values g*16..g*16+15 of its output column o.
      const unsigned short* pBh =
          Whi + (size_t)(nt * 16 + ln) * E_DIM + kb + g * 16;
      const unsigned short* pBl =
          Wlo + (size_t)(nt * 16 + ln) * E_DIM + kb + g * 16;
      v16bf Bh = mk16(*(const u16x8*)(pBh), *(const u16x8*)(pBh + 8));
      v16bf Bl = mk16(*(const u16x8*)(pBl), *(const u16x8*)(pBl + 8));
      acc[i] = __builtin_amdgcn_wmma_f32_16x16x32_bf16(
          false, Ah, false, Bh, (short)0, acc[i], false, false);
      acc[i] = __builtin_amdgcn_wmma_f32_16x16x32_bf16(
          false, Ah, false, Bl, (short)0, acc[i], false, false);
      acc[i] = __builtin_amdgcn_wmma_f32_16x16x32_bf16(
          false, Al, false, Bh, (short)0, acc[i], false, false);
    }
  }

  // ---- Phase 3: bias + interleaved store ----
  // C/D layout: element v of acc = row (mt*16 + g*8 + v), col ln of the tile.
  const size_t rowbase = (size_t)8 * (size_t)blk;
#pragma unroll
  for (int i = 0; i < 4; ++i) {
    const int nt   = ng + 4 * i;
    const int o    = nt * 16 + ln;   // output channel 0..255
    const int half = o >> 7;         // y1 vs y2 -> feat-row parity
    const int oc   = o & 127;
    const float bv = bias[o];
#pragma unroll
    for (int v = 0; v < 8; ++v) {
      const int lr = mt * 16 + g * 8 + v;
      const size_t frow = rowbase + (size_t)(2 * (lr >> 3) + half);
      __builtin_nontemporal_store(
          acc[i][v] + bv,
          out + frow * (size_t)(C_DIM * E_DIM) + (size_t)((lr & 7) * E_DIM + oc));
    }
  }
}

// ---------------------------------------------------------------------------
extern "C" void kernel_launch(void* const* d_in, const int* in_sizes, int n_in,
                              void* d_out, int out_size, void* d_ws,
                              size_t ws_size, hipStream_t stream) {
  (void)n_in; (void)out_size; (void)ws_size;
  const float* feat = (const float*)d_in[0];
  const float* W    = (const float*)d_in[1];
  const float* b    = (const float*)d_in[2];
  float* out        = (float*)d_out;

  unsigned short* Whi = (unsigned short*)d_ws;
  unsigned short* Wlo = Whi + (size_t)O_DIM * E_DIM;  // 32768 elements each

  wprep_kernel<<<(O_DIM * E_DIM + 255) / 256, 256, 0, stream>>>(W, Whi, Wlo);

  const int nfeat  = in_sizes[0] / (C_DIM * E_DIM);  // 131072
  const int rows   = (nfeat / 2) * C_DIM;            // 524288 A-rows
  const int blocks = rows / 32;                      // 16384 blocks
  decompose_gemm_kernel<<<blocks, 256, 0, stream>>>(feat, Whi, Wlo, b, out);
}
